// GQAttention_40896678593187
// MI455X (gfx1250) — matrix operand order repaired and checked by reading, not compiled, not run on hardware
//
#include <hip/hip_runtime.h>
#include <hip/hip_bf16.h>

// ---------------- Types ----------------
typedef __bf16 bf16;
typedef __bf16 bf16x16 __attribute__((ext_vector_type(16)));
typedef __bf16 bf16x8  __attribute__((ext_vector_type(8)));
typedef float  floatx8 __attribute__((ext_vector_type(8)));
typedef unsigned int uint32x4 __attribute__((ext_vector_type(4)));
typedef int int32x4 __attribute__((ext_vector_type(4)));
typedef int int32x8 __attribute__((ext_vector_type(8)));

#define ASG __attribute__((address_space(1)))
#define ASL __attribute__((address_space(3)))

#define WMMA_BF16(a, b, c) \
    __builtin_amdgcn_wmma_f32_16x16x32_bf16(false, (a), false, (b), (short)0, (c), false, false)

#if __has_builtin(__builtin_amdgcn_global_load_async_to_lds_b128) && \
    __has_builtin(__builtin_amdgcn_s_wait_asynccnt)
#define HAVE_ASYNC_LDS 1
#else
#define HAVE_ASYNC_LDS 0
#endif

#if __has_builtin(__builtin_amdgcn_tensor_load_to_lds) && \
    __has_builtin(__builtin_amdgcn_s_wait_tensorcnt)
#define HAVE_TDM 1
#else
#define HAVE_TDM 0
#endif

// ---------------- Problem constants ----------------
constexpr int Bsz  = 16;
constexpr int Tseq = 512;
constexpr int Wdim = 2048;
constexpr int Hh   = 32;
constexpr int KVh  = 8;
constexpr int Dh   = 64;
constexpr int Gg   = Hh / KVh;       // 4
constexpr int HD   = Hh * Dh;        // 2048
constexpr int KVD  = KVh * Dh;       // 512

// ---------------- fp32 -> bf16 conversion ----------------
__global__ void f32_to_bf16_kernel(const float* __restrict__ src,
                                   bf16* __restrict__ dst, int n) {
    int i = blockIdx.x * blockDim.x + threadIdx.x;
    int stride = gridDim.x * blockDim.x;
    for (; i < n; i += stride) dst[i] = (bf16)src[i];
}

// ---------------- 16-lane (DPP row) max reduction, VALU only -------------
template <int CTRL>
__device__ __forceinline__ float dpp_max_step(float x) {
    int s = __builtin_amdgcn_update_dpp(
        0, __builtin_bit_cast(int, x), CTRL, 0xf, 0xf, true);
    return fmaxf(x, __builtin_bit_cast(float, s));
}
__device__ __forceinline__ float row16_max(float x) {
    x = dpp_max_step<0x161>(x);   // ROW_XMASK 1
    x = dpp_max_step<0x162>(x);   // ROW_XMASK 2
    x = dpp_max_step<0x164>(x);   // ROW_XMASK 4
    x = dpp_max_step<0x168>(x);   // ROW_XMASK 8
    return x;
}

#if HAVE_TDM
// ---------------- TDM 2D tile load: global (row stride) -> padded LDS -----
// D# layout per CDNA5 ISA 08_async_tensor §8.3-8.5. data_size=1 (2 bytes).
// pad_int: every (1<<pad_int) 8B-units of a row, insert (pad_amt+1) DWORDs.
__device__ __forceinline__ void tdm_load_2d_bf16(
    const bf16* gsrc, unsigned lds_off, unsigned tile_w, unsigned tile_h,
    unsigned stride_elems, unsigned pad_int, unsigned pad_amt) {
    unsigned long long ga = (unsigned long long)(size_t)gsrc;
    uint32x4 g0;
    g0.x = 1u;                                  // count=1 (valid descriptor)
    g0.y = lds_off;                             // LDS byte address
    g0.z = (unsigned)ga;                        // global_addr[31:0]
    g0.w = (unsigned)(ga >> 32) | (2u << 30);   // global_addr[56:32] | type=2
    int32x8 g1;
    g1[0] = (int)((1u << 16) |                  // data_size = 2 bytes
                  (1u << 20) |                  // pad_enable
                  (pad_int << 22) | (pad_amt << 25));
    g1[1] = (int)((tile_w & 0xffffu) << 16);            // tensor_dim0[15:0]
    g1[2] = (int)((tile_w >> 16) | ((tile_h & 0xffffu) << 16)); // dim0hi|dim1lo
    g1[3] = (int)((tile_h >> 16) | (tile_w << 16));     // dim1hi | tile_dim0
    g1[4] = (int)tile_h;                                // tile_dim1 (dim2=0)
    g1[5] = (int)stride_elems;                          // tensor_dim0_stride lo
    g1[6] = 0;                                          // stride hi | dim1_stride
    g1[7] = 0;
    int32x4 z4 = {};
#if __clang_major__ >= 23
    int32x8 z8 = {};
    __builtin_amdgcn_tensor_load_to_lds(g0, g1, z4, z4, z8, 0);
#else
    __builtin_amdgcn_tensor_load_to_lds(g0, g1, z4, z4, 0);
#endif
}
#endif

// ---------------- bf16 WMMA GEMM: C[M,N] = A[M,K] * B[K,N] ----------------
// Workgroup tile 128x64, 8 waves (4x2), each wave 32x32 (2x2 WMMA frags).
// LDS staging uses GLOBAL_LOAD_ASYNC_TO_LDS_B128 (ASYNCcnt) when available.
template <typename OutT>
__global__ __launch_bounds__(256)
void gemm_bf16_wmma(const bf16* __restrict__ A, const bf16* __restrict__ Bm,
                    OutT* __restrict__ C, int M, int N, int K) {
    constexpr int BM = 128, BN = 64, BK = 32;
    constexpr int LSA = BK + 8;   // 40 elems (80B stride, 16B multiple)
    constexpr int LSB = BN + 8;   // 72 elems (144B stride, 16B multiple)
    __shared__ bf16 As[BM][LSA];
    __shared__ bf16 Bs[BK][LSB];

    const int tid  = threadIdx.x;
    const int lane = tid & 31;
    const int wave = tid >> 5;
    const int wm = wave >> 1, wn = wave & 1;      // 4x2 wave grid
    const int m0 = blockIdx.x * BM, n0 = blockIdx.y * BN;
    const int row16 = lane & 15, half = lane >> 4;

    floatx8 acc[2][2] = {};

    for (int k0 = 0; k0 < K; k0 += BK) {
#if HAVE_ASYNC_LDS
        // async global -> LDS staging (no VGPR bounce), tracked by ASYNCcnt
        #pragma unroll
        for (int it = 0; it < 2; ++it) {
            int idx = tid + it * 256;
            int r = idx >> 2, s = (idx & 3) * 8;
            __builtin_amdgcn_global_load_async_to_lds_b128(
                (ASG int32x4*)&A[(size_t)(m0 + r) * K + k0 + s],
                (ASL int32x4*)&As[r][s], 0, 0);
        }
        {
            int r = tid >> 3, s = (tid & 7) * 8;
            __builtin_amdgcn_global_load_async_to_lds_b128(
                (ASG int32x4*)&Bm[(size_t)(k0 + r) * N + n0 + s],
                (ASL int32x4*)&Bs[r][s], 0, 0);
        }
#else
        #pragma unroll
        for (int it = 0; it < 2; ++it) {
            int idx = tid + it * 256;
            int r = idx >> 2, s = (idx & 3) * 8;
            *(bf16x8*)&As[r][s] =
                *(const bf16x8*)&A[(size_t)(m0 + r) * K + k0 + s];
        }
        {
            int r = tid >> 3, s = (tid & 7) * 8;
            *(bf16x8*)&Bs[r][s] =
                *(const bf16x8*)&Bm[(size_t)(k0 + r) * N + n0 + s];
        }
#endif
        // prefetch next K-chunk while this one is staged (global_prefetch_b8)
        if (k0 + BK < K) {
            #pragma unroll
            for (int it = 0; it < 2; ++it) {
                int idx = tid + it * 256;
                int r = idx >> 2, s = (idx & 3) * 8;
                __builtin_prefetch(&A[(size_t)(m0 + r) * K + k0 + BK + s], 0, 1);
            }
            {
                int r = tid >> 3, s = (tid & 7) * 8;
                __builtin_prefetch(&Bm[(size_t)(k0 + BK + r) * N + n0 + s], 0, 1);
            }
        }
#if HAVE_ASYNC_LDS
        __builtin_amdgcn_s_wait_asynccnt(0);
#endif
        __syncthreads();

        // A fragments (16x32 bf16, ISA A-layout)
        bf16x16 af[2];
        #pragma unroll
        for (int mi = 0; mi < 2; ++mi) {
            int row = wm * 32 + mi * 16 + row16;
            #pragma unroll
            for (int v = 0; v < 8; ++v) {
                int kk = ((v & 4) << 2) + half * 8 + ((v & 3) << 1);
                af[mi][2 * v]     = As[row][kk];
                af[mi][2 * v + 1] = As[row][kk + 1];
            }
        }
        // B fragments (32x16 bf16, ISA B-layout)
        bf16x16 bfv[2];
        #pragma unroll
        for (int ni = 0; ni < 2; ++ni) {
            int col = wn * 32 + ni * 16 + row16;
            #pragma unroll
            for (int e = 0; e < 16; ++e)
                bfv[ni][e] = Bs[half * 8 + e + (e & 8)][col];
        }
        #pragma unroll
        for (int mi = 0; mi < 2; ++mi)
            #pragma unroll
            for (int ni = 0; ni < 2; ++ni)
                acc[mi][ni] = WMMA_BF16(af[mi], bfv[ni], acc[mi][ni]);
        __syncthreads();
    }

    // store (C-layout: row = base + 8*half + r, col = base + (lane&15))
    #pragma unroll
    for (int mi = 0; mi < 2; ++mi)
        #pragma unroll
        for (int ni = 0; ni < 2; ++ni) {
            int colg = n0 + wn * 32 + ni * 16 + row16;
            #pragma unroll
            for (int r = 0; r < 8; ++r) {
                int rowg = m0 + wm * 32 + mi * 16 + half * 8 + r;
                C[(size_t)rowg * N + colg] = (OutT)acc[mi][ni][r];
            }
        }
}

// ---------------- RoPE (in-place, one thread per (b*t, head) row) ---------
__global__ void rope_kernel(bf16* __restrict__ x, int nheads) {
    int idx = blockIdx.x * blockDim.x + threadIdx.x;
    int total = Bsz * Tseq * nheads;
    if (idx >= total) return;
    int h = idx % nheads;
    int bt = idx / nheads;
    int t = bt % Tseq;
    bf16* p = x + ((size_t)bt * nheads + h) * Dh;
    float in[Dh];
    #pragma unroll
    for (int d = 0; d < Dh; ++d) in[d] = (float)p[d];
    #pragma unroll
    for (int i = 0; i < Dh / 2; ++i) {
        float freq = __powf(10000.0f, -(2.0f * i) / (float)Dh);
        float ang = (float)t * freq;
        float s = __sinf(ang), c = __cosf(ang);
        float x1 = in[2 * i], x2 = in[2 * i + 1];
        p[i]          = (bf16)(x1 * c - x2 * s);
        p[i + Dh / 2] = (bf16)(x1 * s + x2 * c);
    }
}

// ---------------- Flash-style GQA attention ----------------
// grid.x = B*H, grid.y = T/64; 128 threads = 4 waves, one 16-row Q tile/wave.
// K/V chunk staging via Tensor Data Mover (TENSORcnt) when available.
__global__ __launch_bounds__(128)
void attn_kernel(const bf16* __restrict__ q, const bf16* __restrict__ k,
                 const bf16* __restrict__ v, bf16* __restrict__ o) {
    constexpr int SC  = 32;        // keys per chunk
    constexpr int LSK = Dh + 8;    // 72 elems, 144B stride (16B multiple)
    __shared__ bf16 Ks[SC][LSK];
    __shared__ bf16 Vs[SC][LSK];
    __shared__ bf16 Ps[4][16][SC]; // per-wave P bounce (C-layout -> A-layout)

    const int tid  = threadIdx.x;
    const int lane = tid & 31;
    const int wave = tid >> 5;
    const int bh = blockIdx.x;
    const int b = bh / Hh, h = bh % Hh, kvh = h / Gg;
    const int t0 = blockIdx.y * 64 + wave * 16;
    const int row16 = lane & 15, half = lane >> 4;
    const float scale = 0.125f;   // 1/sqrt(64)

    // Q fragments: 16x64 row tile -> two 16x32 A-fragments (loaded once)
    bf16x16 qf[2];
    {
        const bf16* qp = q + ((size_t)(b * Tseq + t0 + row16) * Hh + h) * Dh;
        #pragma unroll
        for (int kc = 0; kc < 2; ++kc)
            #pragma unroll
            for (int v2 = 0; v2 < 8; ++v2) {
                int dd = kc * 32 + ((v2 & 4) << 2) + half * 8 + ((v2 & 3) << 1);
                qf[kc][2 * v2]     = qp[dd];
                qf[kc][2 * v2 + 1] = qp[dd + 1];
            }
    }

    // all-ones B fragment: row sums of P via one extra WMMA
    bf16x16 onesb;
    #pragma unroll
    for (int e = 0; e < 16; ++e) onesb[e] = (bf16)1.0f;

    floatx8 oacc[4] = {};            // 4 d-segments of 16
    float mrow[8], lrow[8];
    #pragma unroll
    for (int r = 0; r < 8; ++r) { mrow[r] = -1e30f; lrow[r] = 0.0f; }

    for (int s0 = 0; s0 < Tseq; s0 += SC) {
        __syncthreads();             // prior iter's Ks/Vs/Ps reads done
        const bf16* kp = k + ((size_t)(b * Tseq + s0) * KVh + kvh) * Dh;
        const bf16* vp = v + ((size_t)(b * Tseq + s0) * KVh + kvh) * Dh;
#if HAVE_TDM
        // Tensor Data Mover: 2 DMA tile loads issued by wave 0.
        // 32x64 bf16 tile, row stride KVD elems; LDS pad 16B per 128B row
        // (pad_interval=4 -> 16x8B, pad_amount=3 -> 4 DWORDs) => 144B stride.
        if (wave == 0) {
            tdm_load_2d_bf16(kp, (unsigned)(size_t)&Ks[0][0],
                             Dh, SC, KVD, 4u, 3u);
            tdm_load_2d_bf16(vp, (unsigned)(size_t)&Vs[0][0],
                             Dh, SC, KVD, 4u, 3u);
            __builtin_amdgcn_s_wait_tensorcnt(0);
        }
#elif HAVE_ASYNC_LDS
        #pragma unroll
        for (int it = 0; it < 2; ++it) {
            int idx = tid + it * 128;
            int r = idx >> 3, s = (idx & 7) * 8;
            __builtin_amdgcn_global_load_async_to_lds_b128(
                (ASG int32x4*)&kp[(size_t)r * KVD + s],
                (ASL int32x4*)&Ks[r][s], 0, 0);
            __builtin_amdgcn_global_load_async_to_lds_b128(
                (ASG int32x4*)&vp[(size_t)r * KVD + s],
                (ASL int32x4*)&Vs[r][s], 0, 0);
        }
        __builtin_amdgcn_s_wait_asynccnt(0);
#else
        #pragma unroll
        for (int it = 0; it < 2; ++it) {
            int idx = tid + it * 128;
            int r = idx >> 3, s = (idx & 7) * 8;
            *(bf16x8*)&Ks[r][s] = *(const bf16x8*)&kp[(size_t)r * KVD + s];
            *(bf16x8*)&Vs[r][s] = *(const bf16x8*)&vp[(size_t)r * KVD + s];
        }
#endif
        __syncthreads();

        // scores 16x32 = two 16x16 frags, each 2 WMMAs over K=64
        floatx8 sc[2] = {};
        #pragma unroll
        for (int ns = 0; ns < 2; ++ns) {
            #pragma unroll
            for (int kc = 0; kc < 2; ++kc) {
                bf16x16 bfr;   // B = K^T: (k=d, n=s_local)
                #pragma unroll
                for (int e = 0; e < 16; ++e)
                    bfr[e] = Ks[ns * 16 + row16][kc * 32 + half * 8 + e + (e & 8)];
                sc[ns] = WMMA_BF16(qf[kc], bfr, sc[ns]);
            }
            #pragma unroll
            for (int r = 0; r < 8; ++r) sc[ns][r] *= scale;
        }

        // online softmax: row max via DPP ROW_XMASK (no LDS traffic)
        float alpha[8];
        #pragma unroll
        for (int r = 0; r < 8; ++r) {
            float mloc = row16_max(fmaxf(sc[0][r], sc[1][r]));
            float mnew = fmaxf(mrow[r], mloc);
            alpha[r] = __expf(mrow[r] - mnew);
            mrow[r] = mnew;
            #pragma unroll
            for (int ns = 0; ns < 2; ++ns)
                sc[ns][r] = __expf(sc[ns][r] - mnew);
        }

        // P: C-layout -> LDS -> A-layout bf16 fragment
        #pragma unroll
        for (int ns = 0; ns < 2; ++ns)
            #pragma unroll
            for (int r = 0; r < 8; ++r)
                Ps[wave][half * 8 + r][ns * 16 + row16] = (bf16)sc[ns][r];
        __syncthreads();
        bf16x16 pf;
        #pragma unroll
        for (int v2 = 0; v2 < 8; ++v2) {
            int kk = ((v2 & 4) << 2) + half * 8 + ((v2 & 3) << 1);
            pf[2 * v2]     = Ps[wave][row16][kk];
            pf[2 * v2 + 1] = Ps[wave][row16][kk + 1];
        }

        // row sums of P via WMMA against all-ones B (C-layout matches rows)
        {
            floatx8 zero = {};
            floatx8 sums = WMMA_BF16(pf, onesb, zero);
            #pragma unroll
            for (int r = 0; r < 8; ++r)
                lrow[r] = lrow[r] * alpha[r] + sums[r];
        }

        // rescale O, then O += P @ V  (4 WMMAs over d-segments)
        #pragma unroll
        for (int ds = 0; ds < 4; ++ds) {
            #pragma unroll
            for (int r = 0; r < 8; ++r) oacc[ds][r] *= alpha[r];
            bf16x16 vfr;   // B = V chunk: (k=s_local, n=d)
            #pragma unroll
            for (int e = 0; e < 16; ++e)
                vfr[e] = Vs[half * 8 + e + (e & 8)][ds * 16 + row16];
            oacc[ds] = WMMA_BF16(pf, vfr, oacc[ds]);
        }
    }

    // epilogue: normalize rows, store bf16 [B*T, H*D]
    bf16* op = o + ((size_t)(b * Tseq + t0) * Hh + h) * Dh;
    #pragma unroll
    for (int ds = 0; ds < 4; ++ds)
        #pragma unroll
        for (int r = 0; r < 8; ++r) {
            int rr = half * 8 + r;
            float val = oacc[ds][r] / lrow[r];
            op[(size_t)rr * HD + ds * 16 + row16] = (bf16)val;
        }
}

// ---------------- Host launch ----------------
extern "C" void kernel_launch(void* const* d_in, const int* in_sizes, int n_in,
                              void* d_out, int out_size, void* d_ws, size_t ws_size,
                              hipStream_t stream) {
    (void)in_sizes; (void)n_in; (void)out_size; (void)ws_size;
    const float* x  = (const float*)d_in[0];
    const float* wq = (const float*)d_in[1];
    const float* wk = (const float*)d_in[2];
    const float* wv = (const float*)d_in[3];
    const float* wo = (const float*)d_in[4];
    float* out = (float*)d_out;

    char* ws = (char*)d_ws;
    size_t off = 0;
    auto alloc = [&](size_t elems) {
        bf16* p = (bf16*)(ws + off);
        off += ((elems * sizeof(bf16) + 255) / 256) * 256;
        return p;
    };
    const size_t nX = (size_t)Bsz * Tseq * Wdim;
    bf16* xb  = alloc(nX);
    bf16* wqb = alloc((size_t)Wdim * HD);
    bf16* wkb = alloc((size_t)Wdim * KVD);
    bf16* wvb = alloc((size_t)Wdim * KVD);
    bf16* wob = alloc((size_t)HD * Wdim);
    bf16* qb  = alloc((size_t)Bsz * Tseq * HD);
    bf16* kb  = alloc((size_t)Bsz * Tseq * KVD);
    bf16* vb  = alloc((size_t)Bsz * Tseq * KVD);
    bf16* ob  = xb;   // x no longer needed after projections: reuse

    auto conv = [&](const float* s, bf16* d, size_t n) {
        int blocks = (int)((n + 255) / 256);
        if (blocks > 4096) blocks = 4096;
        f32_to_bf16_kernel<<<blocks, 256, 0, stream>>>(s, d, (int)n);
    };
    conv(x,  xb,  nX);
    conv(wq, wqb, (size_t)Wdim * HD);
    conv(wk, wkb, (size_t)Wdim * KVD);
    conv(wv, wvb, (size_t)Wdim * KVD);
    conv(wo, wob, (size_t)HD * Wdim);

    const int M = Bsz * Tseq;   // 8192
    gemm_bf16_wmma<bf16><<<dim3(M / 128, HD / 64),  256, 0, stream>>>(xb, wqb, qb, M, HD,  Wdim);
    gemm_bf16_wmma<bf16><<<dim3(M / 128, KVD / 64), 256, 0, stream>>>(xb, wkb, kb, M, KVD, Wdim);
    gemm_bf16_wmma<bf16><<<dim3(M / 128, KVD / 64), 256, 0, stream>>>(xb, wvb, vb, M, KVD, Wdim);

    rope_kernel<<<(Bsz * Tseq * Hh  + 127) / 128, 128, 0, stream>>>(qb, Hh);
    rope_kernel<<<(Bsz * Tseq * KVh + 127) / 128, 128, 0, stream>>>(kb, KVh);

    attn_kernel<<<dim3(Bsz * Hh, Tseq / 64), 128, 0, stream>>>(qb, kb, vb, ob);

    gemm_bf16_wmma<float><<<dim3(M / 128, Wdim / 64), 256, 0, stream>>>(ob, wob, out, M, Wdim, Wdim);
}
